// Model_19292993094061
// MI455X (gfx1250) — matrix operand order repaired and checked by reading, not compiled, not run on hardware
//
#include <hip/hip_runtime.h>
#include <hip/hip_bf16.h>

typedef __attribute__((ext_vector_type(16))) _Float16 v16h;
typedef __attribute__((ext_vector_type(8)))  _Float16 v8h;
typedef __attribute__((ext_vector_type(8)))  float    v8f;

#define DIM 64
#define NG  16384
#define NI  16384
#define NTM 131072
#define NTT 65536

__device__ __forceinline__ int lower_bound_i(const int* __restrict__ seg, int n, int key) {
    int lo = 0, hi = n;
    while (lo < hi) {
        int mid = (lo + hi) >> 1;
        if (seg[mid] < key) lo = mid + 1; else hi = mid;
    }
    return lo;
}

// ---- weight convert + transpose: W[K,N] f32 (row-major) -> WT[N,K] f16 ----
__global__ void conv_wt(const float* __restrict__ W, _Float16* __restrict__ WT, int K, int N) {
    int idx = blockIdx.x * blockDim.x + threadIdx.x;
    if (idx >= K * N) return;
    int n = idx / K, k = idx - n * K;
    WT[idx] = (_Float16)W[(size_t)k * N + n];
}

// ---- users: segment mean + concat + u_dat l2norm. grid=G, block=64 ----
__global__ void build_user_inputs(const float* __restrict__ user_tab,
                                  const float* __restrict__ group_dat_tab,
                                  const int* __restrict__ group_ids,
                                  const int* __restrict__ member_ids,
                                  const int* __restrict__ member_seg,
                                  _Float16* __restrict__ Xu,
                                  float* __restrict__ u_dat_out) {
    int g = blockIdx.x;
    int d = threadIdx.x;           // 0..63
    int lo = lower_bound_i(member_seg, NTM, g);
    int hi = lower_bound_i(member_seg, NTM, g + 1);
    float s = 0.f;
    for (int j = lo; j < hi; ++j)
        s += user_tab[(size_t)member_ids[j] * DIM + d];
    float mean = s / fmaxf((float)(hi - lo), 1.f);
    float gd = group_dat_tab[(size_t)group_ids[g] * DIM + d];
    Xu[(size_t)g * 128 + d]        = (_Float16)mean;
    Xu[(size_t)g * 128 + DIM + d]  = (_Float16)gd;
    __shared__ float red[64];
    red[d] = gd * gd;
    __syncthreads();
    for (int st = 32; st > 0; st >>= 1) { if (d < st) red[d] += red[d + st]; __syncthreads(); }
    u_dat_out[(size_t)g * DIM + d] = gd / fmaxf(sqrtf(red[0]), 1e-12f);
}

// ---- items: tag segment mean, build 3 stacked 192-wide rows, idat l2norm ----
__global__ void build_item_inputs(const float* __restrict__ item_tab,
                                  const float* __restrict__ tag_tab,
                                  const float* __restrict__ item_dat_tab,
                                  const int* __restrict__ item_ids,
                                  const int* __restrict__ tag_ids,
                                  const int* __restrict__ tag_seg,
                                  _Float16* __restrict__ Xi,
                                  float* __restrict__ idat_out) {
    int i = blockIdx.x;
    int d = threadIdx.x;
    int lo = lower_bound_i(tag_seg, NTT, i);
    int hi = lower_bound_i(tag_seg, NTT, i + 1);
    float s = 0.f;
    for (int j = lo; j < hi; ++j)
        s += tag_tab[(size_t)tag_ids[j] * DIM + d];
    float tmean = s / fmaxf((float)(hi - lo), 1.f);
    int id = item_ids[i];
    float e   = item_tab[(size_t)id * DIM + d];
    float dat = item_dat_tab[(size_t)id * DIM + d];
    size_t r0 = (size_t)i * 192;                 // full
    size_t r1 = (size_t)(NI + i) * 192;          // only_id
    size_t r2 = (size_t)(2 * NI + i) * 192;      // only_tag
    Xi[r0 + d] = (_Float16)e;    Xi[r0 + 64 + d] = (_Float16)tmean; Xi[r0 + 128 + d] = (_Float16)dat;
    Xi[r1 + d] = (_Float16)e;    Xi[r1 + 64 + d] = (_Float16)0.f;   Xi[r1 + 128 + d] = (_Float16)dat;
    Xi[r2 + d] = (_Float16)0.f;  Xi[r2 + 64 + d] = (_Float16)tmean; Xi[r2 + 128 + d] = (_Float16)dat;
    __shared__ float red[64];
    red[d] = dat * dat;
    __syncthreads();
    for (int st = 32; st > 0; st >>= 1) { if (d < st) red[d] += red[d + st]; __syncthreads(); }
    idat_out[(size_t)i * DIM + d] = dat / fmaxf(sqrtf(red[0]), 1e-12f);
}

// ---- WMMA GEMM: one wave per block, 16 rows x 64 cols (4 N-tiles) per wave ----
// X: [M,K] f16 row-major. WT: [N,K] f16 row-major (B rows = output columns).
// A frag (16x32, contiguous k order used here): lanes 0-15 hold K kk..kk+15; lanes 16-31 kk+16..kk+31.
// B frag (KxN, same k order as A):       lanes 0-15 hold K kk..kk+15; lanes 16-31 kk+16..kk+31.
__device__ __forceinline__ void wmma_tile4(const _Float16* __restrict__ ap,
                                           const _Float16* __restrict__ bp0,
                                           int K, int bstride, v8f acc[4]) {
    const int lane = threadIdx.x & 31;
    const int hs = lane >> 4;
    for (int kk = 0; kk < K; kk += 32) {
        v16h a = *(const v16h*)(ap + kk + hs * 16);
#pragma unroll
        for (int t = 0; t < 4; ++t) {
            v16h b = *(const v16h*)(bp0 + (size_t)t * bstride + kk + hs * 16);
            acc[t] = __builtin_amdgcn_wmma_f32_16x16x32_f16(
                false, a, false, b, (short)0, acc[t], false, false);
        }
    }
}

__global__ void wmma_gemm_bias_relu_h(const _Float16* __restrict__ X,
                                      const _Float16* __restrict__ WT,
                                      const float* __restrict__ bias,
                                      _Float16* __restrict__ H,
                                      int K, int N) {
    const int lane = threadIdx.x & 31;
    const int hs = lane >> 4, l = lane & 15;
    const int n0 = blockIdx.x * 64;                  // 4 N-tiles
    const int m0 = blockIdx.y * 16;
    const _Float16* ap  = X  + (size_t)(m0 + l) * K;
    const _Float16* bp0 = WT + (size_t)(n0 + l) * K; // tile t adds 16*K
    v8f acc[4] = {{}, {}, {}, {}};
    wmma_tile4(ap, bp0, K, 16 * K, acc);
#pragma unroll
    for (int t = 0; t < 4; ++t) {
        int col = n0 + t * 16 + l;
        float bv = bias[col];
#pragma unroll
        for (int r = 0; r < 8; ++r) {
            int row = m0 + hs * 8 + r;
            float v = acc[t][r] + bv;
            v = v > 0.f ? v : 0.f;
            H[(size_t)row * N + col] = (_Float16)v;
        }
    }
}

__global__ void wmma_gemm_bias_f(const _Float16* __restrict__ X,
                                 const _Float16* __restrict__ WT,
                                 const float* __restrict__ bias,
                                 float* __restrict__ Y,
                                 int K, int N) {
    const int lane = threadIdx.x & 31;
    const int hs = lane >> 4, l = lane & 15;
    const int n0 = blockIdx.x * 64;
    const int m0 = blockIdx.y * 16;
    const _Float16* ap  = X  + (size_t)(m0 + l) * K;
    const _Float16* bp0 = WT + (size_t)(n0 + l) * K;
    v8f acc[4] = {{}, {}, {}, {}};
    wmma_tile4(ap, bp0, K, 16 * K, acc);
#pragma unroll
    for (int t = 0; t < 4; ++t) {
        int col = n0 + t * 16 + l;
        float bv = bias[col];
#pragma unroll
        for (int r = 0; r < 8; ++r) {
            int row = m0 + hs * 8 + r;
            Y[(size_t)row * N + col] = acc[t][r] + bv;
        }
    }
}

// ---- row-wise l2 normalize, rows of 64 floats. grid=rows, block=64 ----
__global__ void l2norm_rows(const float* __restrict__ Y, float* __restrict__ out) {
    int r = blockIdx.x, d = threadIdx.x;
    float v = Y[(size_t)r * DIM + d];
    __shared__ float red[64];
    red[d] = v * v;
    __syncthreads();
    for (int st = 32; st > 0; st >>= 1) { if (d < st) red[d] += red[d + st]; __syncthreads(); }
    out[(size_t)r * DIM + d] = v / fmaxf(sqrtf(red[0]), 1e-12f);
}

extern "C" void kernel_launch(void* const* d_in, const int* in_sizes, int n_in,
                              void* d_out, int out_size, void* d_ws, size_t ws_size,
                              hipStream_t stream) {
    const float* user_tab      = (const float*)d_in[0];
    const float* item_tab      = (const float*)d_in[1];
    const float* tag_tab       = (const float*)d_in[2];
    const float* group_dat_tab = (const float*)d_in[3];
    const float* item_dat_tab  = (const float*)d_in[4];
    const float* uW1 = (const float*)d_in[5];
    const float* ub1 = (const float*)d_in[6];
    const float* uW2 = (const float*)d_in[7];
    const float* ub2 = (const float*)d_in[8];
    const float* iW1 = (const float*)d_in[9];
    const float* ib1 = (const float*)d_in[10];
    const float* iW2 = (const float*)d_in[11];
    const float* ib2 = (const float*)d_in[12];
    const int* group_ids  = (const int*)d_in[13];
    const int* member_ids = (const int*)d_in[14];
    const int* member_seg = (const int*)d_in[15];
    const int* item_ids   = (const int*)d_in[16];
    const int* tag_ids    = (const int*)d_in[17];
    const int* tag_seg    = (const int*)d_in[18];
    float* out = (float*)d_out;

    // workspace carve-out
    char* ws = (char*)d_ws;
    size_t off = 0;
    auto carve = [&](size_t bytes) -> char* {
        off = (off + 255) & ~(size_t)255;
        char* p = ws + off;
        off += bytes;
        return p;
    };
    _Float16* uW1T = (_Float16*)carve((size_t)256 * 128 * 2);
    _Float16* uW2T = (_Float16*)carve((size_t)64  * 256 * 2);
    _Float16* iW1T = (_Float16*)carve((size_t)256 * 192 * 2);
    _Float16* iW2T = (_Float16*)carve((size_t)64  * 256 * 2);
    _Float16* Xu   = (_Float16*)carve((size_t)NG * 128 * 2);
    _Float16* Hu   = (_Float16*)carve((size_t)NG * 256 * 2);
    float*    Yu   = (float*)   carve((size_t)NG * 64 * 4);
    _Float16* Xi   = (_Float16*)carve((size_t)3 * NI * 192 * 2);
    _Float16* Hi   = (_Float16*)carve((size_t)3 * NI * 256 * 2);
    float*    Yi   = (float*)   carve((size_t)3 * NI * 64 * 4);

    const size_t SEC = (size_t)NG * DIM;   // one output section (== NI*DIM)

    // weights -> f16 transposed
    conv_wt<<<(128 * 256 + 255) / 256, 256, 0, stream>>>(uW1, uW1T, 128, 256);
    conv_wt<<<(256 * 64  + 255) / 256, 256, 0, stream>>>(uW2, uW2T, 256, 64);
    conv_wt<<<(192 * 256 + 255) / 256, 256, 0, stream>>>(iW1, iW1T, 192, 256);
    conv_wt<<<(256 * 64  + 255) / 256, 256, 0, stream>>>(iW2, iW2T, 256, 64);

    // gather + segment means + direct dat outputs
    build_user_inputs<<<NG, 64, 0, stream>>>(user_tab, group_dat_tab, group_ids,
                                             member_ids, member_seg, Xu, out + 1 * SEC);
    build_item_inputs<<<NI, 64, 0, stream>>>(item_tab, tag_tab, item_dat_tab,
                                             item_ids, tag_ids, tag_seg, Xi, out + 3 * SEC);

    // user tower: [G,128] -> relu[G,256] -> [G,64] -> l2norm
    wmma_gemm_bias_relu_h<<<dim3(256 / 64, NG / 16), 32, 0, stream>>>(Xu, uW1T, ub1, Hu, 128, 256);
    wmma_gemm_bias_f    <<<dim3(64  / 64, NG / 16), 32, 0, stream>>>(Hu, uW2T, ub2, Yu, 256, 64);
    l2norm_rows<<<NG, 64, 0, stream>>>(Yu, out + 0 * SEC);

    // item tower (3 variants stacked): [3I,192] -> relu[3I,256] -> [3I,64]
    wmma_gemm_bias_relu_h<<<dim3(256 / 64, 3 * NI / 16), 32, 0, stream>>>(Xi, iW1T, ib1, Hi, 192, 256);
    wmma_gemm_bias_f    <<<dim3(64  / 64, 3 * NI / 16), 32, 0, stream>>>(Hi, iW2T, ib2, Yi, 256, 64);
    l2norm_rows<<<NI, 64, 0, stream>>>(Yi,                        out + 2 * SEC); // full
    l2norm_rows<<<NI, 64, 0, stream>>>(Yi + (size_t)NI * DIM,     out + 4 * SEC); // only_id
    l2norm_rows<<<NI, 64, 0, stream>>>(Yi + (size_t)2 * NI * DIM, out + 5 * SEC); // only_tag
}
